// Mamba_global_36326833389646
// MI455X (gfx1250) — compile-verified, matrix operand order checked by reading, not run on hardware
//
#include <hip/hip_runtime.h>
#include <hip/hip_bf16.h>

typedef __attribute__((ext_vector_type(16))) _Float16 v16h;
typedef __attribute__((ext_vector_type(8)))  _Float16 v8h;
typedef __attribute__((ext_vector_type(8)))  float    v8f;
typedef __attribute__((ext_vector_type(4)))  unsigned int u32x4;
typedef __attribute__((ext_vector_type(8)))  int          i32x8;
typedef __attribute__((ext_vector_type(4)))  int          i32x4;

#define LTOK   4097
#define LPAD   4104   // row stride (multiple of 8 -> 16B/32B aligned rows)
#define HW     4096
#define DIM_   256
#define DINNER 512
#define DSTATE 16
#define DTRANK 16
#define BATCH  4
#define NT_L   257    // ceil(4097/16)

// K index for 16-bit WMMA A/B fragments: half j (0..15) within the 8 VGPRs.
__device__ __forceinline__ int kmap(int lane, int j) {
  return ((j >> 3) << 4) + ((lane >> 4) << 3) + (j & 7);
}

__device__ __forceinline__ float sigmoidf_(float x) { return 1.0f / (1.0f + __expf(-x)); }
__device__ __forceinline__ float siluf_(float x)    { return x * sigmoidf_(x); }
__device__ __forceinline__ float softplusf_(float x){ return (x > 20.f) ? x : log1pf(__expf(x)); }

// ---- Tensor Data Mover: 2-D tile (t0 x t1 elements of 4B) global -> LDS ----
// Descriptor per CDNA5 ISA section 8.3/8.4. 6-arg builtin (clang-23 / therock).
__device__ __forceinline__ void tdm_load_2d(unsigned lds_off, const void* gaddr,
                                            unsigned n0, unsigned n1,
                                            unsigned t0, unsigned t1, unsigned stride0)
{
  unsigned long long ga = (unsigned long long)gaddr;
  u32x4 g0;
  g0[0] = 1u;                                        // count=1, user mode
  g0[1] = lds_off;                                   // LDS byte address
  g0[2] = (unsigned)(ga & 0xffffffffu);              // global_addr[31:0]
  g0[3] = (unsigned)((ga >> 32) & 0x01ffffffu) | (2u << 30); // addr[56:32] | type=2
  i32x8 g1;
  g1[0] = (int)(2u << 16);                                   // data_size = 4 bytes
  g1[1] = (int)((n0 & 0xffffu) << 16);                       // tensor_dim0 lo
  g1[2] = (int)(((n0 >> 16) & 0xffffu) | ((n1 & 0xffffu) << 16)); // dim0 hi | dim1 lo
  g1[3] = (int)(((n1 >> 16) & 0xffffu) | ((t0 & 0xffffu) << 16)); // dim1 hi | tile_dim0
  g1[4] = (int)(t1 & 0xffffu);                               // tile_dim1 (tile_dim2=0)
  g1[5] = (int)stride0;                                      // tensor_dim0_stride lo
  g1[6] = 0;
  g1[7] = 0;
  i32x4 gz4 = {0, 0, 0, 0};
  i32x8 gz8 = {0, 0, 0, 0, 0, 0, 0, 0};
  __builtin_amdgcn_tensor_load_to_lds(g0, g1, gz4, gz4, gz8, 0);
}

// ---------------- Kernel 1: token build + in_proj GEMM ----------------
// Block = one (b, et) pair, 8 lt tiles; A weight tile staged in LDS via TDM.
__global__ void k_inproj(const float* __restrict__ x, const float* __restrict__ gtok,
                         const float* __restrict__ W, _Float16* __restrict__ xc,
                         _Float16* __restrict__ z)
{
  __shared__ float As[16 * DIM_];                 // 16 KB
  const int lane = threadIdx.x & 31;
  const int wave = threadIdx.x >> 5;
  int grp = blockIdx.x % 33;
  int be  = blockIdx.x / 33;                      // 0..255
  int b   = be >> 6;
  int et  = be & 63;

  if (wave == 0) {
    tdm_load_2d((unsigned)(size_t)(void*)As, W + (size_t)et * 16 * DIM_,
                DIM_, 16, DIM_, 16, DIM_);
    __builtin_amdgcn_s_wait_tensorcnt(0);
  }
  __syncthreads();

  int lt = grp * 8 + wave;
  if (lt >= NT_L) return;                         // wave-uniform, after barrier

  const float* xb = x + (size_t)b * DIM_ * HW;
  const int lcol = lt * 16 + (lane & 15);
  // branch-free B source: per-lane base pointer / stride / scale
  const float* bp; int bstride; float bscale;
  if (lcol == 0)       { bp = gtok;            bstride = 1;  bscale = 1.f; }
  else if (lcol <= HW) { bp = xb + (lcol - 1); bstride = HW; bscale = 1.f; }
  else                 { bp = xb;              bstride = 0;  bscale = 0.f; }

  v8f acc = {};
  for (int kk = 0; kk < 8; ++kk) {
    int k0 = kk * 32;
    v16h a, bf;
#pragma unroll
    for (int jj = 0; jj < 8; ++jj) {
      int j = jj * 2;
      int k = k0 + kmap(lane, j);
      float2 w2 = *(const float2*)&As[(lane & 15) * DIM_ + k];
      a[j] = (_Float16)w2.x; a[j + 1] = (_Float16)w2.y;
    }
#pragma unroll
    for (int j = 0; j < 16; ++j) {
      int d = k0 + kmap(lane, j);
      bf[j] = (_Float16)(bp[(size_t)d * bstride] * bscale);
    }
    acc = __builtin_amdgcn_wmma_f32_16x16x32_f16(false, a, false, bf, (short)0, acc, false, false);
  }
  // ---- store epilogue: block-uniform dest select, single per-lane guard ----
  _Float16* dst = (et < 32) ? xc : z;             // block-uniform
  int ebase = ((et & 31) << 4) + ((lane >> 4) << 3);
  if (lcol < LTOK) {
    _Float16* p = dst + ((size_t)b * DINNER + ebase) * LPAD + lcol;
#pragma unroll
    for (int r = 0; r < 8; ++r) p[(size_t)r * LPAD] = (_Float16)acc[r];
  }
}

// ---------------- Kernel 2: causal depthwise conv (k=4) + bias + SiLU ----------------
__global__ void k_conv(const _Float16* __restrict__ xc, const float* __restrict__ cw,
                       const float* __restrict__ cb, _Float16* __restrict__ xcs)
{
  size_t i = (size_t)blockIdx.x * 256 + threadIdx.x;
  const size_t TOT = (size_t)BATCH * DINNER * LPAD;
  if (i >= TOT) return;
  int l  = (int)(i % LPAD);
  int ce = (int)(i / LPAD);           // b*512 + e
  if (l >= LTOK) return;
  int e  = ce % DINNER;
  float s = cb[e];
#pragma unroll
  for (int tq = 0; tq < 4; ++tq) {
    int lt = l - 3 + tq;
    if (lt >= 0) s += cw[e * 4 + tq] * (float)xc[(size_t)ce * LPAD + lt];
  }
  xcs[(size_t)ce * LPAD + l] = (_Float16)siluf_(s);
}

// ---------------- Kernel 3: x_proj GEMM (M=48, K=512) -> x_dbl f32 [b,48,LPAD] ----------------
__global__ void k_xdbl(const _Float16* __restrict__ xcs, const float* __restrict__ W,
                       float* __restrict__ xdbl)
{
  const int lane = threadIdx.x & 31;
  int t = blockIdx.x * 8 + (threadIdx.x >> 5);
  const int TILES = BATCH * 3 * NT_L;
  if (t >= TILES) return;
  int b  = t / (3 * NT_L);
  int r0 = t % (3 * NT_L);
  int jt = r0 / NT_L;
  int lt = r0 % NT_L;
  const int row  = jt * 16 + (lane & 15);
  const int lcol = lt * 16 + (lane & 15);
  const int lc   = (lcol < LTOK) ? lcol : (LTOK - 1);   // clamp, select later
  const bool lvalid = (lcol < LTOK);
  const _Float16* xb = xcs + (size_t)b * DINNER * LPAD;
  v8f acc = {};
  for (int kk = 0; kk < 16; ++kk) {
    int k0 = kk * 32;
    v16h a, bf;
#pragma unroll
    for (int jj = 0; jj < 8; ++jj) {
      int j = jj * 2;
      int k = k0 + kmap(lane, j);
      float2 w2 = *(const float2*)(W + (size_t)row * DINNER + k);
      a[j] = (_Float16)w2.x; a[j + 1] = (_Float16)w2.y;
    }
#pragma unroll
    for (int j = 0; j < 16; ++j) {
      int k = k0 + kmap(lane, j);
      _Float16 hv = xb[(size_t)k * LPAD + lc];
      bf[j] = lvalid ? hv : (_Float16)0.f;
    }
    acc = __builtin_amdgcn_wmma_f32_16x16x32_f16(false, a, false, bf, (short)0, acc, false, false);
  }
  if (lvalid) {
    float* p = xdbl + ((size_t)b * 48 + jt * 16 + ((lane >> 4) << 3)) * LPAD + lcol;
#pragma unroll
    for (int r = 0; r < 8; ++r) p[(size_t)r * LPAD] = acc[r];
  }
}

// ---------------- Kernel 4: dt_proj GEMM (K=16 padded to 32) + softplus -> dt f16 ----------------
__global__ void k_dt(const float* __restrict__ xdbl, const float* __restrict__ W,
                     const float* __restrict__ bias, _Float16* __restrict__ dt)
{
  const int lane = threadIdx.x & 31;
  int t = blockIdx.x * 8 + (threadIdx.x >> 5);
  int b  = t / (32 * NT_L);
  int r0 = t % (32 * NT_L);
  int dtile = r0 / NT_L;
  int lt    = r0 % NT_L;
  int row   = dtile * 16 + (lane & 15);
  int lcol  = lt * 16 + (lane & 15);
  const int lc = (lcol < LTOK) ? lcol : (LTOK - 1);
  const bool lvalid = (lcol < LTOK);
  const float* xdb = xdbl + (size_t)b * 48 * LPAD;
  v16h a, bf;
#pragma unroll
  for (int jj = 0; jj < 4; ++jj) {               // j = 0..7 -> K = 0..15 (real K)
    int j = jj * 2;
    int k = kmap(lane, j);
    float2 w2 = *(const float2*)(W + (size_t)row * DTRANK + k);
    a[j] = (_Float16)w2.x; a[j + 1] = (_Float16)w2.y;
  }
#pragma unroll
  for (int j = 0; j < 8; ++j) {
    int k = kmap(lane, j);
    float v = xdb[(size_t)k * LPAD + lc];
    bf[j] = lvalid ? (_Float16)v : (_Float16)0.f;
  }
#pragma unroll
  for (int j = 8; j < 16; ++j) { a[j] = (_Float16)0.f; bf[j] = (_Float16)0.f; } // K=16..31 pad
  v8f acc = {};
  acc = __builtin_amdgcn_wmma_f32_16x16x32_f16(false, a, false, bf, (short)0, acc, false, false);
  int d0 = dtile * 16 + ((lane >> 4) << 3);
  if (lvalid) {
    _Float16* p = dt + ((size_t)b * DINNER + d0) * LPAD + lcol;
#pragma unroll
    for (int r = 0; r < 8; ++r)
      p[(size_t)r * LPAD] = (_Float16)softplusf_(acc[r] + bias[d0 + r]);
  }
}

// ---------------- Kernel 5: selective scan, 8 timesteps per vector-load block ----------------
__global__ void k_scan(const _Float16* __restrict__ xcs, const _Float16* __restrict__ dt,
                       const float* __restrict__ xdbl, const _Float16* __restrict__ zg,
                       const float* __restrict__ A_log, const float* __restrict__ Dskip,
                       _Float16* __restrict__ y)
{
  const int lane = threadIdx.x & 31;
  int w = blockIdx.x * 8 + (threadIdx.x >> 5);  // 0..1023
  int c = w * 2 + (lane >> 4);                  // b*512 + d   (2 channels per wave)
  int b = c >> 9;
  int d = c & 511;
  int n = lane & 15;
  float Av = -__expf(A_log[d * DSTATE + n]);
  float Dd = Dskip[d];
  const _Float16* xrow  = xcs + (size_t)c * LPAD;
  const _Float16* dtrow = dt  + (size_t)c * LPAD;
  const _Float16* zrow  = zg  + (size_t)c * LPAD;
  const float* Brow = xdbl + ((size_t)b * 48 + DTRANK + n) * LPAD;
  const float* Crow = xdbl + ((size_t)b * 48 + DTRANK + DSTATE + n) * LPAD;
  _Float16* yrow = y + (size_t)c * LPAD;
  float h = 0.f;
  for (int l0 = 0; l0 < HW; l0 += 8) {          // l = 0..4095
    v8h dt8 = *(const v8h*)(dtrow + l0);
    v8h x8  = *(const v8h*)(xrow + l0);
    v8h z8  = *(const v8h*)(zrow + l0);
    v8f B8  = *(const v8f*)(Brow + l0);
    v8f C8  = *(const v8f*)(Crow + l0);
    v8h yo;
#pragma unroll
    for (int q = 0; q < 8; ++q) {
      float dtv = (float)dt8[q];
      float xv  = (float)x8[q];
      float dA  = __expf(dtv * Av);
      h = dA * h + (dtv * xv) * B8[q];
      float p = h * C8[q];
      p += __shfl_xor(p, 1);
      p += __shfl_xor(p, 2);
      p += __shfl_xor(p, 4);
      p += __shfl_xor(p, 8);
      float zv = (float)z8[q];
      yo[q] = (_Float16)((p + xv * Dd) * siluf_(zv));
    }
    if (n == 0) *(v8h*)(yrow + l0) = yo;
  }
  {                                             // epilogue l = 4096
    int l = HW;
    float dtv = (float)dtrow[l];
    float xv  = (float)xrow[l];
    float dA  = __expf(dtv * Av);
    h = dA * h + (dtv * xv) * Brow[l];
    float p = h * Crow[l];
    p += __shfl_xor(p, 1);
    p += __shfl_xor(p, 2);
    p += __shfl_xor(p, 4);
    p += __shfl_xor(p, 8);
    if (n == 0) {
      float zv = (float)zrow[l];
      yrow[l] = (_Float16)((p + xv * Dd) * siluf_(zv));
    }
  }
}

// ---------------- Kernel 6: out_proj GEMM -> d_out (B,256,64,64) f32 ----------------
// Block = one (b, ct), 8 lt tiles; A tile (16x512 f32 = 32KB) staged via TDM.
__global__ void k_outproj(const _Float16* __restrict__ y, const float* __restrict__ W,
                          float* __restrict__ out)
{
  __shared__ float As[16 * DINNER];               // 32 KB
  const int lane = threadIdx.x & 31;
  const int wave = threadIdx.x >> 5;
  int grp = blockIdx.x & 31;
  int be  = blockIdx.x >> 5;                      // 0..63
  int b   = be >> 4;
  int ct  = be & 15;

  if (wave == 0) {
    tdm_load_2d((unsigned)(size_t)(void*)As, W + (size_t)ct * 16 * DINNER,
                DINNER, 16, DINNER, 16, DINNER);
    __builtin_amdgcn_s_wait_tensorcnt(0);
  }
  __syncthreads();

  int lt = grp * 8 + wave;                        // 0..255, always valid
  int lcol = 1 + lt * 16 + (lane & 15);           // skip global token
  const _Float16* yb = y + (size_t)b * DINNER * LPAD;
  v8f acc = {};
  for (int kk = 0; kk < 16; ++kk) {
    int k0 = kk * 32;
    v16h a, bf;
#pragma unroll
    for (int jj = 0; jj < 8; ++jj) {
      int j = jj * 2;
      int k = k0 + kmap(lane, j);
      float2 w2 = *(const float2*)&As[(lane & 15) * DINNER + k];
      a[j] = (_Float16)w2.x; a[j + 1] = (_Float16)w2.y;
    }
#pragma unroll
    for (int j = 0; j < 16; ++j) {
      int k = k0 + kmap(lane, j);
      bf[j] = yb[(size_t)k * LPAD + lcol];
    }
    acc = __builtin_amdgcn_wmma_f32_16x16x32_f16(false, a, false, bf, (short)0, acc, false, false);
  }
  float* p = out + ((size_t)b * DIM_ + ct * 16 + ((lane >> 4) << 3)) * HW + (lcol - 1);
#pragma unroll
  for (int r = 0; r < 8; ++r) p[(size_t)r * HW] = acc[r];
}

extern "C" void kernel_launch(void* const* d_in, const int* in_sizes, int n_in,
                              void* d_out, int out_size, void* d_ws, size_t ws_size,
                              hipStream_t stream) {
  (void)in_sizes; (void)n_in; (void)out_size; (void)ws_size;
  const float* x        = (const float*)d_in[0];
  const float* gtok     = (const float*)d_in[1];
  const float* in_w     = (const float*)d_in[2];
  const float* conv_w   = (const float*)d_in[3];
  const float* conv_b   = (const float*)d_in[4];
  const float* xproj_w  = (const float*)d_in[5];
  const float* dtproj_w = (const float*)d_in[6];
  const float* dtproj_b = (const float*)d_in[7];
  const float* A_log    = (const float*)d_in[8];
  const float* Dskip    = (const float*)d_in[9];
  const float* out_w    = (const float*)d_in[10];
  float* out = (float*)d_out;

  char* ws = (char*)d_ws;
  const size_t NE = (size_t)BATCH * DINNER * LPAD;  // 8,404,992 elems per [b,512,LPAD] buffer
  _Float16* ws_xc   = (_Float16*)(ws);              // f16
  _Float16* ws_z    = (_Float16*)(ws + NE * 2);     // f16
  _Float16* ws_xcs  = (_Float16*)(ws + NE * 4);     // f16 (conv+silu output)
  _Float16* ws_dt   = (_Float16*)(ws + NE * 6);     // f16 (softplus dt)
  float*    ws_xdbl = (float*)   (ws + NE * 8);     // f32 [b,48,LPAD]
  _Float16* ws_y    = ws_xc;                        // reuse xc buffer for gated y

  k_inproj <<<8448, 256, 0, stream>>>(x, gtok, in_w, ws_xc, ws_z);
  k_conv   <<<(int)((NE + 255) / 256), 256, 0, stream>>>(ws_xc, conv_w, conv_b, ws_xcs);
  k_xdbl   <<<386, 256, 0, stream>>>(ws_xcs, xproj_w, ws_xdbl);
  k_dt     <<<4112, 256, 0, stream>>>(ws_xdbl, dtproj_w, dtproj_b, ws_dt);
  k_scan   <<<128, 256, 0, stream>>>(ws_xcs, ws_dt, ws_xdbl, ws_z, A_log, Dskip, ws_y);
  k_outproj<<<2048, 256, 0, stream>>>(ws_y, out_w, out);
}
